// TorchLMHeadGRPO_59090160058697
// MI455X (gfx1250) — matrix operand order repaired and checked by reading, not compiled, not run on hardware
//
#include <hip/hip_runtime.h>
#include <math.h>

// Problem constants (from reference): B=4, T=1024, H=2048, V=32000
#define NTOK 4096
#define HDIM 2048
#define VOC  32000
#define NSPLIT 10
#define NCOLS_SPLIT (VOC / NSPLIT)    // 3200 vocab cols per split
#define NTILE 128                      // N tile (vocab cols) per workgroup step
#define NTILES (NCOLS_SPLIT / NTILE)   // 25 tiles per split
#define MBLK 128                       // tokens per workgroup
#define KCH 64                         // K chunk staged in LDS
#define CPT (HDIM / KCH)               // 32 chunks per tile
#define LDSPAD 8                       // bf16 pad to avoid bank conflicts

#define BETA_C 0.1f
#define EPS_LO 0.2f
#define EPS_HI 0.2f

typedef unsigned short u16;
typedef unsigned long long u64;
typedef __attribute__((ext_vector_type(16))) __bf16 v16bf;
typedef __attribute__((ext_vector_type(8)))  __bf16 v8bf;
typedef __attribute__((ext_vector_type(8)))  float  v8f;

union ABfrag { v16bf v; v8bf h[2]; };

__device__ __forceinline__ u16 f2bf(float f) {
  unsigned u = __float_as_uint(f);
  u += 0x7fffu + ((u >> 16) & 1u);   // round-to-nearest-even
  return (u16)(u >> 16);
}

// reductions over the 16-lane half-wave (wave32)
__device__ __forceinline__ float hred16(float v) {
  v += __shfl_xor(v, 1, 16);
  v += __shfl_xor(v, 2, 16);
  v += __shfl_xor(v, 4, 16);
  v += __shfl_xor(v, 8, 16);
  return v;
}
__device__ __forceinline__ float hmax16(float v) {
  v = fmaxf(v, __shfl_xor(v, 1, 16));
  v = fmaxf(v, __shfl_xor(v, 2, 16));
  v = fmaxf(v, __shfl_xor(v, 4, 16));
  v = fmaxf(v, __shfl_xor(v, 8, 16));
  return v;
}

// CDNA5 async global->LDS copy, 16 bytes per lane (ASYNCcnt-tracked)
#define ASYNC_LD16(ldsoff, gptr)                                        \
  asm volatile("global_load_async_to_lds_b128 %0, %1, off"              \
               :: "v"(ldsoff), "v"(gptr) : "memory")

__device__ __forceinline__ void wait_async0() {
#if __has_builtin(__builtin_amdgcn_s_wait_asynccnt)
  __builtin_amdgcn_s_wait_asynccnt(0);
#else
  asm volatile("s_wait_asynccnt 0x0" ::: "memory");
#endif
}

// ---------------- Phase 0: f32 -> bf16 convert ----------------
__global__ void cvt_bf16_kernel(const float* __restrict__ src,
                                u16* __restrict__ dst, long n4) {
  long i = (long)blockIdx.x * blockDim.x + threadIdx.x;
  if (i >= n4) return;
  float4 f = ((const float4*)src)[i];
  u64 p = (u64)f2bf(f.x) | ((u64)f2bf(f.y) << 16) |
          ((u64)f2bf(f.z) << 32) | ((u64)f2bf(f.w) << 48);
  ((u64*)dst)[i] = p;
}

// One K-chunk of the 128x128 tile: 16 WMMAs per wave. Schedule is pinned per
// k-step: 18 DS reads (A frag + all 8 B frags) then 8 WMMAs, so the waitcnt
// pass can overlap WMMA i with the still-in-flight loads of frags i+1..7.
__device__ __forceinline__ void compute_chunk(
    const u16 (*__restrict__ Alb)[KCH + LDSPAD],
    const u16 (*__restrict__ Blb)[KCH + LDSPAD],
    int mband, int lm, int h, v8f* acc) {
#pragma unroll
  for (int k2 = 0; k2 < KCH; k2 += 32) {
    // A fragment (16x32 bf16): lanes 0-15 rows m, K {0..7,16..23};
    // lanes 16-31 rows m, K {8..15,24..31}
    ABfrag a;
    a.h[0] = *(const v8bf*)&Alb[mband + lm][k2 + h * 8];
    a.h[1] = *(const v8bf*)&Alb[mband + lm][k2 + 16 + h * 8];
    // B fragments (32x16 bf16): col n=lm, lanes 0-15 K {0..7,16..23}, 16-31 K {8..15,24..31}
    ABfrag bb[8];
#pragma unroll
    for (int j = 0; j < 8; ++j) {
      bb[j].h[0] = *(const v8bf*)&Blb[j * 16 + lm][k2 + h * 8];
      bb[j].h[1] = *(const v8bf*)&Blb[j * 16 + lm][k2 + 16 + h * 8];
    }
#pragma unroll
    for (int j = 0; j < 8; ++j)
      acc[j] = __builtin_amdgcn_wmma_f32_16x16x32_bf16(
          false, a.v, false, bb[j].v, (short)0, acc[j], false, false);
#if __has_builtin(__builtin_amdgcn_sched_group_barrier)
    __builtin_amdgcn_sched_group_barrier(0x100, 18, 0);  // 18 DS reads
    __builtin_amdgcn_sched_group_barrier(0x008, 8, 0);   // 8 WMMAs
#endif
  }
}

// ---------------- Phase 1: bf16 WMMA GEMM + online logsumexp ----------------
// grid: (NTOK/MBLK, NSPLIT, 2 gemms), block: 256 threads (8 waves).
// Wave w owns token band [16w,16w+16) x all 128 cols of the current N tile.
// Global->LDS staging uses async-to-LDS with double-buffered K chunks,
// pipelined across N tiles (next tile's chunk 0 prefetched during the last
// chunk of the current tile).
__global__ __launch_bounds__(256)
void gemm_lse_kernel(const u16* __restrict__ A0, const u16* __restrict__ A1,
                     const u16* __restrict__ W0, const u16* __restrict__ W1,
                     const int* __restrict__ ids,
                     float* __restrict__ pmax, float* __restrict__ psum,
                     float* __restrict__ pslog, float* __restrict__ psel) {
  __shared__ u16 Al[2][MBLK][KCH + LDSPAD];
  __shared__ u16 Bl[2][NTILE][KCH + LDSPAD];

  const int tid   = threadIdx.x;
  const int wid   = tid >> 5;
  const int lane  = tid & 31;
  const int h     = lane >> 4;     // half-wave (0/1)
  const int lm    = lane & 15;
  const int tok0  = blockIdx.x * MBLK;
  const int split = blockIdx.y;
  const int gemm  = blockIdx.z;
  const int mband = wid * 16;

  const u16* __restrict__ Abf = gemm ? A1 : A0;
  const u16* __restrict__ Wbf = gemm ? W1 : W0;

  // selected token ids for this lane's 8 rows
  int myid[8];
#pragma unroll
  for (int r = 0; r < 8; ++r) myid[r] = ids[tok0 + mband + h * 8 + r];

  float run_max[8], run_sum[8], run_slog[8], run_sel[8];
#pragma unroll
  for (int r = 0; r < 8; ++r) {
    run_max[r] = -__builtin_inff();
    run_sum[r] = 0.f; run_slog[r] = 0.f; run_sel[r] = 0.f;
  }

  // staging coordinates: thread t -> row t/2, 32-element half t%2 (64B/thread)
  const int ldrow = tid >> 1;
  const int ldh   = (tid & 1) * 32;
  const unsigned alds[2] = {(unsigned)(size_t)&Al[0][ldrow][ldh],
                            (unsigned)(size_t)&Al[1][ldrow][ldh]};
  const unsigned blds[2] = {(unsigned)(size_t)&Bl[0][ldrow][ldh],
                            (unsigned)(size_t)&Bl[1][ldrow][ldh]};

  // global source pointers for linear chunk index c (tiles x K-chunks)
  auto chunk_ptrs = [&](int c, const u16*& ap, const u16*& bp) {
    int t = c / CPT;
    int k = (c - t * CPT) * KCH;
    int nbl = split * NCOLS_SPLIT + t * NTILE;
    ap = Abf + (size_t)(tok0 + ldrow) * HDIM + ldh + k;
    bp = Wbf + (size_t)(nbl + ldrow) * HDIM + ldh + k;
  };
  auto issue_chunk = [&](int c, int buf) {
    const u16 *ap, *bp;
    chunk_ptrs(c, ap, bp);
#pragma unroll
    for (int j = 0; j < 4; ++j) {
      ASYNC_LD16(alds[buf] + (unsigned)(j * 16), ap + j * 8);
      ASYNC_LD16(blds[buf] + (unsigned)(j * 16), bp + j * 8);
    }
    __builtin_prefetch((const void*)(ap + KCH), 0, 1);   // hint chunk c+1
    __builtin_prefetch((const void*)(bp + KCH), 0, 1);
  };

  const int NCHUNK = NTILES * CPT;
  issue_chunk(0, 0);                    // prologue: chunk 0 -> buf 0

  int cidx = 0;
  for (int nt = 0; nt < NTILES; ++nt) {
    const int nb = split * NCOLS_SPLIT + nt * NTILE;   // global col base

    v8f acc[8];
    const v8f zero = {0.f, 0.f, 0.f, 0.f, 0.f, 0.f, 0.f, 0.f};
#pragma unroll
    for (int i = 0; i < 8; ++i) acc[i] = zero;

    for (int kc = 0; kc < HDIM; kc += 2 * KCH) {
      // ---- buffer 0 (even chunk) ----
      wait_async0();
      __syncthreads();                  // all waves' chunk data visible; all
                                        // waves past compute of buf1's prior use
      if (cidx + 1 < NCHUNK) issue_chunk(cidx + 1, 1);
      compute_chunk(Al[0], Bl[0], mband, lm, h, acc);
      ++cidx;
      // ---- buffer 1 (odd chunk) ----
      wait_async0();
      __syncthreads();
      if (cidx + 1 < NCHUNK) issue_chunk(cidx + 1, 0);
      compute_chunk(Al[1], Bl[1], mband, lm, h, acc);
      ++cidx;
    }

    // ---- epilogue: online logsumexp over this 128-col tile, in-register.
    // C layout: element (VGPR r, lane) -> row = mband + h*8 + r, col = i*16+lm
#pragma unroll
    for (int r = 0; r < 8; ++r) {
      float tmax = acc[0][r];
#pragma unroll
      for (int i = 1; i < 8; ++i) tmax = fmaxf(tmax, acc[i][r]);
      tmax = hmax16(tmax);
      float nm = fmaxf(run_max[r], tmax);
      float scale = __expf(run_max[r] - nm);   // exp(-inf)=0 on first tile
      float se = 0.f, sl = 0.f;
#pragma unroll
      for (int i = 0; i < 8; ++i) {
        float v = acc[i][r];
        se += __expf(v - nm);
        sl += v;
      }
      se = hred16(se);
      sl = hred16(sl);
      run_sum[r]  = run_sum[r] * scale + se;
      run_max[r]  = nm;
      run_slog[r] += sl;

      int c = myid[r] - nb;     // selected logit if it lives in this tile
      float sv = 0.f;
      if ((unsigned)c < (unsigned)NTILE) {
#pragma unroll
        for (int i = 0; i < 8; ++i)
          if (((c >> 4) == i) && ((c & 15) == lm)) sv = acc[i][r];
      }
      sv = hred16(sv);
      run_sel[r] += sv;
    }
  }

  // write per-split partials (one lane per half-wave writes its 8 rows)
  if (lm == 0) {
#pragma unroll
    for (int r = 0; r < 8; ++r) {
      int t = tok0 + mband + h * 8 + r;
      size_t o = ((size_t)(gemm * NSPLIT + split)) * NTOK + t;
      pmax[o]  = run_max[r];
      psum[o]  = run_sum[r];
      pslog[o] = run_slog[r];
      psel[o]  = run_sel[r];
    }
  }
}

// ---------------- Phase 2: combine splits -> per-token logp ----------------
__global__ void combine_kernel(const float* __restrict__ pmax,
                               const float* __restrict__ psum,
                               const float* __restrict__ pslog,
                               const float* __restrict__ psel,
                               float* __restrict__ ptl,
                               float* __restrict__ lpm) {
  int idx = blockIdx.x * blockDim.x + threadIdx.x;
  if (idx >= 2 * NTOK) return;
  int g = idx / NTOK, t = idx % NTOK;
  float M = -__builtin_inff();
#pragma unroll
  for (int s = 0; s < NSPLIT; ++s)
    M = fmaxf(M, pmax[((size_t)(g * NSPLIT + s)) * NTOK + t]);
  float S = 0.f, sl = 0.f, sv = 0.f;
#pragma unroll
  for (int s = 0; s < NSPLIT; ++s) {
    size_t o = ((size_t)(g * NSPLIT + s)) * NTOK + t;
    S  += psum[o] * __expf(pmax[o] - M);
    sl += pslog[o];
    sv += psel[o];
  }
  float lse = M + __logf(S);
  ptl[idx] = sv - lse;                        // selected logit - logsumexp
  if (g == 0) lpm[t] = sl * (1.0f / VOC) - lse;  // mean_v(log_probs) per token
}

// ---------------- Phase 3: GRPO loss + 4 scalar outputs ----------------
__global__ __launch_bounds__(256)
void finalize_kernel(const float* __restrict__ ptl, const float* __restrict__ lpm,
                     const float* __restrict__ mask, const float* __restrict__ adv,
                     float* __restrict__ out) {
  float lnum = 0.f, msum = 0.f, psum = 0.f, lsum = 0.f;
  float kln[4] = {0.f, 0.f, 0.f, 0.f}, kld[4] = {0.f, 0.f, 0.f, 0.f};
  for (int t = threadIdx.x; t < NTOK; t += 256) {
    int b = t >> 10;                               // T = 1024
    float p = ptl[t], rp = ptl[NTOK + t], m = mask[t];
    float a = adv[b];
    float c1 = 1.0f;                               // exp(p - stopgrad(p))
    float c2 = fminf(fmaxf(c1, 1.f - EPS_LO), 1.f + EPS_HI);
    float pl = -fminf(c1 * a, c2 * a);
    float d = rp - p;
    float kl = __expf(d) - d - 1.f;
    pl += BETA_C * kl;
    lnum += pl * m; msum += m; psum += p; lsum += lpm[t];
    kln[b] += kl * m; kld[b] += m;
  }
  float vals[12] = {lnum, msum, psum, lsum,
                    kln[0], kln[1], kln[2], kln[3],
                    kld[0], kld[1], kld[2], kld[3]};
#pragma unroll
  for (int q = 0; q < 12; ++q) {
    float v = vals[q];
    v += __shfl_xor(v, 16, 32);
    v += __shfl_xor(v, 8, 32);
    v += __shfl_xor(v, 4, 32);
    v += __shfl_xor(v, 2, 32);
    v += __shfl_xor(v, 1, 32);
    vals[q] = v;
  }
  __shared__ float red[8][12];
  int wid = threadIdx.x >> 5, lane = threadIdx.x & 31;
  if (lane == 0)
#pragma unroll
    for (int q = 0; q < 12; ++q) red[wid][q] = vals[q];
  __syncthreads();
  if (threadIdx.x == 0) {
    float s[12];
#pragma unroll
    for (int q = 0; q < 12; ++q) {
      float acc = 0.f;
      for (int w = 0; w < 8; ++w) acc += red[w][q];
      s[q] = acc;
    }
    out[0] = s[0] / fmaxf(s[1], 1.f);          // loss
    out[1] = s[2] * (1.0f / NTOK);             // per_token_logps.mean()
    out[2] = s[3] * (1.0f / NTOK);             // log_probs.mean()
    float km = 0.f;
    for (int b = 0; b < 4; ++b) km += s[4 + b] / fmaxf(s[8 + b], 1.f);
    out[3] = km * 0.25f;                       // kl metric
  }
}

extern "C" void kernel_launch(void* const* d_in, const int* in_sizes, int n_in,
                              void* d_out, int out_size, void* d_ws, size_t ws_size,
                              hipStream_t stream) {
  const float* x    = (const float*)d_in[0];
  const float* W    = (const float*)d_in[1];
  const float* rW   = (const float*)d_in[2];
  const float* rx   = (const float*)d_in[3];
  const int*   ids  = (const int*)d_in[4];    // per harness: integer -> const int*
  const float* mask = (const float*)d_in[5];
  const float* adv  = (const float*)d_in[6];

  char* ws = (char*)d_ws;                      // needs ~300 MB workspace
  size_t off = 0;
  u16* xb  = (u16*)(ws + off); off += (size_t)NTOK * HDIM * 2;
  u16* rb  = (u16*)(ws + off); off += (size_t)NTOK * HDIM * 2;
  u16* Wb  = (u16*)(ws + off); off += (size_t)VOC * HDIM * 2;
  u16* rWb = (u16*)(ws + off); off += (size_t)VOC * HDIM * 2;
  float* pmax  = (float*)(ws + off); off += (size_t)2 * NSPLIT * NTOK * 4;
  float* psum  = (float*)(ws + off); off += (size_t)2 * NSPLIT * NTOK * 4;
  float* pslog = (float*)(ws + off); off += (size_t)2 * NSPLIT * NTOK * 4;
  float* psel  = (float*)(ws + off); off += (size_t)2 * NSPLIT * NTOK * 4;
  float* ptl   = (float*)(ws + off); off += (size_t)2 * NTOK * 4;
  float* lpm   = (float*)(ws + off); off += (size_t)NTOK * 4;

  long n4a = (long)NTOK * HDIM / 4;
  long n4w = (long)VOC * HDIM / 4;
  cvt_bf16_kernel<<<(unsigned)((n4a + 255) / 256), 256, 0, stream>>>(x, xb, n4a);
  cvt_bf16_kernel<<<(unsigned)((n4a + 255) / 256), 256, 0, stream>>>(rx, rb, n4a);
  cvt_bf16_kernel<<<(unsigned)((n4w + 255) / 256), 256, 0, stream>>>(W, Wb, n4w);
  cvt_bf16_kernel<<<(unsigned)((n4w + 255) / 256), 256, 0, stream>>>(rW, rWb, n4w);

  dim3 grid(NTOK / MBLK, NSPLIT, 2);
  gemm_lse_kernel<<<grid, 256, 0, stream>>>(xb, rb, Wb, rWb, ids,
                                            pmax, psum, pslog, psel);

  combine_kernel<<<(2 * NTOK + 255) / 256, 256, 0, stream>>>(pmax, psum, pslog,
                                                             psel, ptl, lpm);
  finalize_kernel<<<1, 256, 0, stream>>>(ptl, lpm, mask, adv, (float*)d_out);
}
